// ContrastLoss_43585328120141
// MI455X (gfx1250) — matrix operand order repaired and checked by reading, not compiled, not run on hardware
//
#include <hip/hip_runtime.h>
#include <hip/hip_bf16.h>

// ---------------------------------------------------------------------------
// Problem constants (from the reference): B=1024, D=128, C=100, KQ=32768
// ---------------------------------------------------------------------------
#define B_DIM  1024
#define D_DIM  128
#define C_DIM  100
#define KQ_DIM 32768
#define N_DIM  (B_DIM + KQ_DIM)      // 33792
#define NT_TILES (N_DIM / 16)        // 2112 column tiles of 16
#define RT_TILES (B_DIM / 16)        // 64 row tiles of 16

typedef __attribute__((ext_vector_type(16))) __bf16 v16bf;
typedef __attribute__((ext_vector_type(8)))  __bf16 v8bf;
typedef __attribute__((ext_vector_type(8)))  float  v8f;

// ---------------------------------------------------------------------------
// Workspace layout (bytes). All regions 256B-aligned.
//   stats : [0]=maxw, [8..1031]=rowmax, [1032..2055]=rowsum (16 KB reserved)
//   q_hi/q_lo : [B,128]  bf16   (query split hi+lo)
//   lab       : [B,128]  bf16   (labels, K padded to 128; {0,1} exact)
//   dict_hi/lo: [N,128]  bf16   (concat(keys, queue.T) split hi+lo)
//   lcat      : [N,128]  bf16   (concat(labels, queue_label.T), K padded)
// ---------------------------------------------------------------------------
#define WS_STATS_B   16384
#define QSZ_B        (B_DIM * 128 * 2)         // 262144
#define DSZ_B        ((size_t)N_DIM * 128 * 2) // 8650752
#define OFF_QHI      ((size_t)WS_STATS_B)
#define OFF_QLO      (OFF_QHI + QSZ_B)
#define OFF_LAB      (OFF_QLO + QSZ_B)
#define OFF_DHI      (OFF_LAB + QSZ_B)
#define OFF_DLO      (OFF_DHI + DSZ_B)
#define OFF_LCAT     (OFF_DLO + DSZ_B)

// ---------------------------------------------------------------------------
// Fragment loaders from preprocessed bf16 arrays (row stride = 128 elements).
// A 16x32 bf16 (ISA 7.12.2): lane m=lane&15, h=lane>>4;
//   elements a[0..7]  = K (kb + 8h     .. +7)   -> contiguous 16B
//   elements a[8..15] = K (kb + 16 + 8h .. +7)  -> contiguous 16B
// B 32x16 bf16: lane n=lane&15, h=lane>>4;
//   elements b[0..7]  = K (kb + 8h     .. +7), b[8..15] = K (kb + 16 + 8h .. +7)
// ---------------------------------------------------------------------------
__device__ __forceinline__ v16bf load_a_frag(const __bf16* __restrict__ base,
                                             int rowbase, int kb) {
    const int lane = threadIdx.x & 31;
    const int m = lane & 15, h = lane >> 4;
    const __bf16* p = base + (size_t)(rowbase + m) * 128 + kb + 8 * h;
    const v8bf lo8 = *(const v8bf*)p;
    const v8bf hi8 = *(const v8bf*)(p + 16);
    v16bf r;
#pragma unroll
    for (int t = 0; t < 8; ++t) { r[t] = lo8[t]; r[8 + t] = hi8[t]; }
    return r;
}

__device__ __forceinline__ v16bf load_b_frag(const __bf16* __restrict__ base,
                                             int colbase, int kb) {
    const int lane = threadIdx.x & 31;
    const int n = lane & 15, h = lane >> 4;
    const __bf16* p = base + (size_t)(colbase + n) * 128 + kb + 8 * h;
    const v8bf lo8 = *(const v8bf*)p;
    const v8bf hi8 = *(const v8bf*)(p + 16);
    v16bf r;
#pragma unroll
    for (int t = 0; t < 8; ++t) { r[t] = lo8[t]; r[8 + t] = hi8[t]; }
    return r;
}

// sim tile (raw dot product, not yet /T): bf16 hi/lo split GEMM, f32 accum.
__device__ __forceinline__ v8f sim_tile(const v16bf (&ahi)[4], const v16bf (&alo)[4],
                                        const __bf16* __restrict__ dict_hi,
                                        const __bf16* __restrict__ dict_lo,
                                        int colbase) {
    v8f acc = {};
#pragma unroll
    for (int c = 0; c < 4; ++c) {
        const v16bf bhi = load_b_frag(dict_hi, colbase, c * 32);
        const v16bf blo = load_b_frag(dict_lo, colbase, c * 32);
        acc = __builtin_amdgcn_wmma_f32_16x16x32_bf16(false, alo[c], false, bhi,
                                                      (short)0, acc, false, false);
        acc = __builtin_amdgcn_wmma_f32_16x16x32_bf16(false, ahi[c], false, blo,
                                                      (short)0, acc, false, false);
        acc = __builtin_amdgcn_wmma_f32_16x16x32_bf16(false, ahi[c], false, bhi,
                                                      (short)0, acc, false, false);
    }
    return acc;
}

// w tile: labels @ l.T ({0,1} entries -> exact in bf16).
__device__ __forceinline__ v8f w_tile(const v16bf (&alab)[4],
                                      const __bf16* __restrict__ lcat, int colbase) {
    v8f acc = {};
#pragma unroll
    for (int c = 0; c < 4; ++c) {
        const v16bf b = load_b_frag(lcat, colbase, c * 32);
        acc = __builtin_amdgcn_wmma_f32_16x16x32_bf16(false, alab[c], false, b,
                                                      (short)0, acc, false, false);
    }
    return acc;
}

// Float atomic max supporting negatives (int-bits trick).
__device__ __forceinline__ void atomicMaxFloat(float* addr, float val) {
    if (val >= 0.0f)
        atomicMax((int*)addr, __float_as_int(val));
    else
        atomicMin((unsigned int*)addr, __float_as_uint(val));
}

// ---------------------------------------------------------------------------
// Kernel: init workspace stats.
// ---------------------------------------------------------------------------
__global__ void k_init_ws(float* __restrict__ wsf) {
    const int i = blockIdx.x * blockDim.x + threadIdx.x;
    if (i == 0) wsf[0] = 0.0f;
    if (i < B_DIM) {
        wsf[8 + i]         = -3.402823466e38f;  // rowmax = -FLT_MAX
        wsf[8 + B_DIM + i] = 0.0f;              // rowsum = 0
    }
}

// ---------------------------------------------------------------------------
// Kernel: max(w) == max_i nnz(labels_i) (binary labels; diagonal attains max).
// ---------------------------------------------------------------------------
__global__ void k_maxw(const float* __restrict__ labels, float* __restrict__ maxw) {
    const int i = blockIdx.x * blockDim.x + threadIdx.x;
    if (i >= B_DIM) return;
    const float* row = labels + (size_t)i * C_DIM;
    float s = 0.0f;
    for (int c = 0; c < C_DIM; ++c) s += row[c];
    atomicMax((int*)maxw, __float_as_int(s));   // s >= 0 -> int-bits compare OK
}

// ---------------------------------------------------------------------------
// Kernel: row-major preprocessing (query, keys, labels): fully coalesced.
// 131072 threads; thread i -> (row = i>>7, k = i&127).
// ---------------------------------------------------------------------------
__global__ void k_prep_direct(const float* __restrict__ query,
                              const float* __restrict__ keys,
                              const float* __restrict__ labels,
                              __bf16* __restrict__ q_hi, __bf16* __restrict__ q_lo,
                              __bf16* __restrict__ lab,
                              __bf16* __restrict__ dict_hi, __bf16* __restrict__ dict_lo,
                              __bf16* __restrict__ lcat) {
    const int i = blockIdx.x * blockDim.x + threadIdx.x;
    if (i >= B_DIM * 128) return;
    const int row = i >> 7, k = i & 127;

    const float fq = query[i];
    const __bf16 qh = (__bf16)fq;
    q_hi[i] = qh; q_lo[i] = (__bf16)(fq - (float)qh);

    const float fk = keys[i];
    const __bf16 kh = (__bf16)fk;
    dict_hi[i] = kh; dict_lo[i] = (__bf16)(fk - (float)kh);

    const float fl = (k < C_DIM) ? labels[(size_t)row * C_DIM + k] : 0.0f;
    const __bf16 lb = (__bf16)fl;
    lab[i] = lb; lcat[i] = lb;
}

// ---------------------------------------------------------------------------
// Kernel: transpose queue [128,KQ] and queue_label [100,KQ] into [col,128]
// bf16 via LDS tiles (64 cols x 128 K, stride 65 to dodge bank conflicts).
// 512 blocks x 256 threads; block -> 64 columns.
// ---------------------------------------------------------------------------
__global__ void k_prep_queueT(const float* __restrict__ queue,
                              const float* __restrict__ qlabel,
                              __bf16* __restrict__ dict_hi, __bf16* __restrict__ dict_lo,
                              __bf16* __restrict__ lcat) {
    __shared__ float tile[128 * 65];
    const int cbase = blockIdx.x * 64;
    const int tc = threadIdx.x & 63;     // column within tile
    const int tk = threadIdx.x >> 6;     // 0..3

    // ---- queue -> dict_hi / dict_lo ----
    for (int k0 = 0; k0 < 128; k0 += 4) {
        const int k = k0 + tk;
        tile[k * 65 + tc] = queue[(size_t)k * KQ_DIM + cbase + tc];  // coalesced read
    }
    __syncthreads();
    for (int idx = threadIdx.x; idx < 64 * 128; idx += 256) {
        const int c = idx >> 7, k = idx & 127;
        const float f = tile[k * 65 + c];
        const __bf16 h = (__bf16)f;
        const size_t o = (size_t)(B_DIM + cbase + c) * 128 + k;       // coalesced write
        dict_hi[o] = h; dict_lo[o] = (__bf16)(f - (float)h);
    }
    __syncthreads();

    // ---- queue_label -> lcat (K padded 100 -> 128) ----
    for (int k0 = 0; k0 < 128; k0 += 4) {
        const int k = k0 + tk;
        tile[k * 65 + tc] = (k < C_DIM) ? qlabel[(size_t)k * KQ_DIM + cbase + tc] : 0.0f;
    }
    __syncthreads();
    for (int idx = threadIdx.x; idx < 64 * 128; idx += 256) {
        const int c = idx >> 7, k = idx & 127;
        lcat[(size_t)(B_DIM + cbase + c) * 128 + k] = (__bf16)tile[k * 65 + c];
    }
}

// ---------------------------------------------------------------------------
// Kernel: row maxima of sim (= raw_dot / T), unmasked (matches reference).
// grid (24, 64), block 256 (8 waves); 2112/(24*8) = 11 tiles per wave.
// ---------------------------------------------------------------------------
#define AB_TPW 11
__global__ void k_rowmax(const __bf16* __restrict__ q_hi, const __bf16* __restrict__ q_lo,
                         const __bf16* __restrict__ dict_hi, const __bf16* __restrict__ dict_lo,
                         const float* __restrict__ Tp, float* __restrict__ rowmax) {
    const int wave = threadIdx.x >> 5;
    const int lane = threadIdx.x & 31;
    const int rowbase = blockIdx.y * 16;
    const float invT = 1.0f / Tp[0];

    v16bf ahi[4], alo[4];
#pragma unroll
    for (int c = 0; c < 4; ++c) {
        ahi[c] = load_a_frag(q_hi, rowbase, c * 32);
        alo[c] = load_a_frag(q_lo, rowbase, c * 32);
    }

    float mx[8];
#pragma unroll
    for (int r = 0; r < 8; ++r) mx[r] = -3.402823466e38f;

    const int t0 = (blockIdx.x * 8 + wave) * AB_TPW;
    for (int k = 0; k < AB_TPW; ++k) {
        const v8f s = sim_tile(ahi, alo, dict_hi, dict_lo, (t0 + k) * 16);
#pragma unroll
        for (int r = 0; r < 8; ++r) mx[r] = fmaxf(mx[r], s[r] * invT);
    }
    // reduce over the 16 lanes sharing a row (xor masks < 16 stay in half-wave)
#pragma unroll
    for (int r = 0; r < 8; ++r) {
        float v = mx[r];
#pragma unroll
        for (int off = 1; off < 16; off <<= 1) v = fmaxf(v, __shfl_xor(v, off, 32));
        if ((lane & 15) == 0)
            atomicMaxFloat(&rowmax[rowbase + r + 8 * (lane >> 4)], v);
    }
}

// ---------------------------------------------------------------------------
// Kernel: rowsum = sum_j exp(sim - rowmax) * logits_mask (diagonal zeroed).
// ---------------------------------------------------------------------------
__global__ void k_rowsum(const __bf16* __restrict__ q_hi, const __bf16* __restrict__ q_lo,
                         const __bf16* __restrict__ dict_hi, const __bf16* __restrict__ dict_lo,
                         const float* __restrict__ Tp, const float* __restrict__ rowmax,
                         float* __restrict__ rowsum) {
    const int wave = threadIdx.x >> 5;
    const int lane = threadIdx.x & 31;
    const int h = lane >> 4;
    const int rowbase = blockIdx.y * 16;
    const float invT = 1.0f / Tp[0];

    v16bf ahi[4], alo[4];
#pragma unroll
    for (int c = 0; c < 4; ++c) {
        ahi[c] = load_a_frag(q_hi, rowbase, c * 32);
        alo[c] = load_a_frag(q_lo, rowbase, c * 32);
    }

    float rm[8], acc[8];
#pragma unroll
    for (int r = 0; r < 8; ++r) { rm[r] = rowmax[rowbase + r + 8 * h]; acc[r] = 0.0f; }

    const int t0 = (blockIdx.x * 8 + wave) * AB_TPW;
    for (int k = 0; k < AB_TPW; ++k) {
        const int colbase = (t0 + k) * 16;
        const v8f s = sim_tile(ahi, alo, dict_hi, dict_lo, colbase);
        const int col = colbase + (lane & 15);
#pragma unroll
        for (int r = 0; r < 8; ++r) {
            const int row = rowbase + r + 8 * h;
            float e = __expf(s[r] * invT - rm[r]);
            if (col == row) e = 0.0f;           // logits_mask: zero self in keys block
            acc[r] += e;
        }
    }
#pragma unroll
    for (int r = 0; r < 8; ++r) {
        float v = acc[r];
#pragma unroll
        for (int off = 1; off < 16; off <<= 1) v += __shfl_xor(v, off, 32);
        if ((lane & 15) == 0)
            atomicAdd(&rowsum[rowbase + r + 8 * (lane >> 4)], v);
    }
}

// ---------------------------------------------------------------------------
// Kernel: output pass. grid (33, 64), block 256; 2112/(33*8) = 8 tiles/wave.
// loss = -(T/BT) * (w>0) * (sim - rowmax - log(rowsum)) * sqrt(w/maxw)
// ---------------------------------------------------------------------------
#define C_TPW 8
__global__ void k_output(const __bf16* __restrict__ q_hi, const __bf16* __restrict__ q_lo,
                         const __bf16* __restrict__ lab,
                         const __bf16* __restrict__ dict_hi, const __bf16* __restrict__ dict_lo,
                         const __bf16* __restrict__ lcat,
                         const float* __restrict__ Tp, const float* __restrict__ BTp,
                         const float* __restrict__ maxw,
                         const float* __restrict__ rowmax, const float* __restrict__ rowsum,
                         float* __restrict__ out) {
    const int wave = threadIdx.x >> 5;
    const int lane = threadIdx.x & 31;
    const int h = lane >> 4;
    const int rowbase = blockIdx.y * 16;

    const float T = Tp[0];
    const float invT = 1.0f / T;
    const float scale = -T / BTp[0];
    const float invmaxw = 1.0f / maxw[0];

    v16bf ahi[4], alo[4], alab[4];
#pragma unroll
    for (int c = 0; c < 4; ++c) {
        ahi[c]  = load_a_frag(q_hi, rowbase, c * 32);
        alo[c]  = load_a_frag(q_lo, rowbase, c * 32);
        alab[c] = load_a_frag(lab,  rowbase, c * 32);
    }

    float rm[8], lg[8];
#pragma unroll
    for (int r = 0; r < 8; ++r) {
        const int row = rowbase + r + 8 * h;
        rm[r] = rowmax[row];
        lg[r] = logf(rowsum[row]);
    }

    const int t0 = (blockIdx.x * 8 + wave) * C_TPW;
    for (int k = 0; k < C_TPW; ++k) {
        const int colbase = (t0 + k) * 16;
        const v8f s  = sim_tile(ahi, alo, dict_hi, dict_lo, colbase);
        const v8f wv = w_tile(alab, lcat, colbase);
        const int col = colbase + (lane & 15);
#pragma unroll
        for (int r = 0; r < 8; ++r) {
            const int row = rowbase + r + 8 * h;
            const float wt = wv[r];
            const float lp = s[r] * invT - rm[r] - lg[r];
            const float res = (wt > 0.0f) ? scale * lp * sqrtf(wt * invmaxw) : 0.0f;
            out[(size_t)row * N_DIM + col] = res;
        }
    }
}

// ---------------------------------------------------------------------------
// Host entry
// ---------------------------------------------------------------------------
extern "C" void kernel_launch(void* const* d_in, const int* in_sizes, int n_in,
                              void* d_out, int out_size, void* d_ws, size_t ws_size,
                              hipStream_t stream) {
    const float* query  = (const float*)d_in[0];
    const float* keys   = (const float*)d_in[1];
    const float* labels = (const float*)d_in[2];
    const float* queue  = (const float*)d_in[3];
    const float* qlabel = (const float*)d_in[4];
    // d_in[5] = K (compile-time constant here)
    const float* Tp  = (const float*)d_in[6];
    const float* BTp = (const float*)d_in[7];
    float* out = (float*)d_out;

    char* ws = (char*)d_ws;
    float* wsf    = (float*)ws;
    float* maxw   = wsf;               // 1 value (padded)
    float* rowmax = wsf + 8;           // 1024
    float* rowsum = wsf + 8 + B_DIM;   // 1024
    __bf16* q_hi    = (__bf16*)(ws + OFF_QHI);
    __bf16* q_lo    = (__bf16*)(ws + OFF_QLO);
    __bf16* lab     = (__bf16*)(ws + OFF_LAB);
    __bf16* dict_hi = (__bf16*)(ws + OFF_DHI);
    __bf16* dict_lo = (__bf16*)(ws + OFF_DLO);
    __bf16* lcat    = (__bf16*)(ws + OFF_LCAT);

    k_init_ws    <<<dim3(8),   dim3(256), 0, stream>>>(wsf);
    k_maxw       <<<dim3(4),   dim3(256), 0, stream>>>(labels, maxw);
    k_prep_direct<<<dim3(512), dim3(256), 0, stream>>>(query, keys, labels,
                                                       q_hi, q_lo, lab,
                                                       dict_hi, dict_lo, lcat);
    k_prep_queueT<<<dim3(512), dim3(256), 0, stream>>>(queue, qlabel,
                                                       dict_hi, dict_lo, lcat);
    k_rowmax<<<dim3(24, 64), dim3(256), 0, stream>>>(q_hi, q_lo, dict_hi, dict_lo,
                                                     Tp, rowmax);
    k_rowsum<<<dim3(24, 64), dim3(256), 0, stream>>>(q_hi, q_lo, dict_hi, dict_lo,
                                                     Tp, rowmax, rowsum);
    k_output<<<dim3(33, 64), dim3(256), 0, stream>>>(q_hi, q_lo, lab, dict_hi, dict_lo,
                                                     lcat, Tp, BTp, maxw, rowmax, rowsum,
                                                     out);
}
